// SiglipAttention_46909632807231
// MI455X (gfx1250) — matrix operand order repaired and checked by reading, not compiled, not run on hardware
//
#include <hip/hip_runtime.h>
#include <hip/hip_bf16.h>

// ---------------------------------------------------------------------------
// SigLIP attention on MI455X (gfx1250, wave32, WMMA).
// All matmuls on v_wmma_f32_16x16x32_bf16 (f32 accumulate).
// GEMM: LDS-staged, double-buffered, async-to-LDS when the toolchain has it.
// ---------------------------------------------------------------------------

typedef __bf16 bf16;
typedef __attribute__((ext_vector_type(16))) __bf16 v16bf;
typedef __attribute__((ext_vector_type(8)))  __bf16 v8bf;
typedef __attribute__((ext_vector_type(8)))  float  v8f;

#define N_TOK 11664      // B*S = 16*729  (== 729 * 16, exact M tiles)
#define DIM   1152
#define BATCH 16
#define SEQ   729
#define HEADS 16
#define HDIM  72
#define SP    736        // SEQ padded to 46*16
#define DP    96         // HDIM padded to 3*32
#define QTILES 46        // ceil(729/16)
#define KSTEPS (DIM / 32)  // 36
#define ATTN_SCALE 0.117851130197757920f   // 72^-0.5

#if __has_builtin(__builtin_amdgcn_global_load_async_to_lds_b128) && \
    __has_builtin(__builtin_amdgcn_s_wait_asynccnt)
#define USE_ASYNC_LDS 1
// builtin signature: (v4i addrspace(1)* src, v4i addrspace(3)* dst, Ii off, Ii cpol)
typedef __attribute__((vector_size(16))) int v4i_t;
typedef __attribute__((address_space(1))) v4i_t v4i_glob;
typedef __attribute__((address_space(3))) v4i_t v4i_lds;
__device__ __forceinline__ void async_b128(const void* gsrc, void* ldst) {
  __builtin_amdgcn_global_load_async_to_lds_b128(
      (v4i_glob*)(uintptr_t)gsrc,
      (v4i_lds*)(uint32_t)(uintptr_t)ldst, 0, 0);
}
#else
#define USE_ASYNC_LDS 0
#endif

// xor-butterfly across the 16-lane row group (bits 0..3 only -> stays in half)
#define SWZ_XOR(x, k) \
  __int_as_float(__builtin_amdgcn_ds_swizzle(__float_as_int(x), (((k) << 10) | 0x1f)))

__device__ __forceinline__ v16bf cat8(v8bf lo, v8bf hi) {
  return __builtin_shufflevector(lo, hi, 0,1,2,3,4,5,6,7,8,9,10,11,12,13,14,15);
}

// ---------------------------------------------------------------------------
// fp32 -> bf16 elementwise convert
__global__ __launch_bounds__(256) void cvt_bf16_kernel(const float* __restrict__ src,
                                                       bf16* __restrict__ dst, int n) {
  int i = blockIdx.x * 256 + threadIdx.x;
  if (i < n) dst[i] = (bf16)src[i];
}

// fp32 weight [K=1152][N=1152] -> bf16 transposed [N][K] (coalesced writes)
__global__ __launch_bounds__(256) void cvt_wT_kernel(const float* __restrict__ w,
                                                     bf16* __restrict__ wt) {
  int i = blockIdx.x * 256 + threadIdx.x;      // i = n*DIM + k
  int k = i % DIM, n = i / DIM;
  wt[i] = (bf16)w[(size_t)k * DIM + n];
}

// ---------------------------------------------------------------------------
// WMMA GEMM: C[M=11664][N=1152] = A_bf16[M][K=1152] @ Wt_bf16[N][K]^T + bias
// Block: 256 thr = 8 waves as (mW 0..1) x (nW 0..3). Block tile 64(M) x 128(N).
// Wave tile: 2x2 16x16 WMMA tiles (4 WMMAs per 32-wide K-step).
// K-loop double-buffers 64x32 A and 128x32 B tiles in LDS (async when avail).
// Epilogue modes:
//   0: f32 row-major to out                            (final projection)
//   1: bf16 scatter to padded head layout [bh][SP][DP] (Q, K)
//   2: bf16 scatter to transposed layout  [bh][DP][SP] (V)
__global__ __launch_bounds__(256)
void gemm_wmma_kernel(const bf16* __restrict__ A, const bf16* __restrict__ Wt,
                      const float* __restrict__ bias, void* __restrict__ out,
                      int mode) {
  __shared__ __align__(32) bf16 ldsA[2][64][32];    //  8 KB
  __shared__ __align__(32) bf16 ldsB[2][128][32];   // 16 KB

  const int t    = threadIdx.x;
  const int lane = t & 31;
  const int wave = t >> 5;
  const int mW   = wave >> 2;                 // 0..1
  const int nW   = wave & 3;                  // 0..3
  const int half = lane >> 4;
  const int l16  = lane & 15;

  const int mBase = blockIdx.x * 64;          // token row base (may overrun, clamp)
  const int nBase = blockIdx.y * 128;         // output col base (exact: 9*128=1152)

  // cooperative staging indices (per thread, fixed)
  const int aRow = t >> 2, aChunk = (t & 3) * 8;    // A: 64 rows x 4 x 8 elems
  const int bRow = t >> 1, bChunk = (t & 1) * 16;   // B: 128 rows x 2 x 16 elems
  const int aSrcRow = min(mBase + aRow, N_TOK - 1); // clamp OOB rows (stores guarded)
  const bf16* aSrc = A  + (size_t)aSrcRow * DIM + aChunk;
  const bf16* bSrc = Wt + (size_t)(nBase + bRow) * DIM + bChunk;

#if USE_ASYNC_LDS
#define STAGE(kk, bufi)                                                        \
  do {                                                                         \
    async_b128(aSrc + (kk),     &ldsA[bufi][aRow][aChunk]);                    \
    async_b128(bSrc + (kk),     &ldsB[bufi][bRow][bChunk]);                    \
    async_b128(bSrc + (kk) + 8, &ldsB[bufi][bRow][bChunk + 8]);                \
  } while (0)
#define ASYNC_WAIT() __builtin_amdgcn_s_wait_asynccnt(0)
#else
#define STAGE(kk, bufi)                                                        \
  do {                                                                         \
    v8bf  av = *(const v8bf*)(aSrc + (kk));                                    \
    v16bf bv = *(const v16bf*)(bSrc + (kk));                                   \
    *(v8bf*)&ldsA[bufi][aRow][aChunk]  = av;                                   \
    *(v16bf*)&ldsB[bufi][bRow][bChunk] = bv;                                   \
  } while (0)
#define ASYNC_WAIT() do {} while (0)
#endif

  v8f acc[2][2];
#pragma unroll
  for (int mt = 0; mt < 2; ++mt)
#pragma unroll
    for (int nt = 0; nt < 2; ++nt) acc[mt][nt] = (v8f){};

  STAGE(0, 0);                                  // prologue stage of k-step 0
  int buf = 0;
  for (int ks = 0; ks < KSTEPS; ++ks) {
    ASYNC_WAIT();                               // own async (into buf) done
    __syncthreads();                            // all waves staged / done reading
    if (ks + 1 < KSTEPS) STAGE((ks + 1) * 32, buf ^ 1);

    // build fragments from LDS
    v16bf af[2], bfv[2];
#pragma unroll
    for (int mt = 0; mt < 2; ++mt) {
      const int r = (mW * 2 + mt) * 16 + l16;
      v8bf lo = *(const v8bf*)&ldsA[buf][r][half ? 8  : 0];
      v8bf hi = *(const v8bf*)&ldsA[buf][r][half ? 24 : 16];
      af[mt] = cat8(lo, hi);
    }
#pragma unroll
    for (int nt = 0; nt < 2; ++nt) {
      const int r = (nW * 2 + nt) * 16 + l16;
      bfv[nt] = cat8(*(const v8bf*)&ldsB[buf][r][half ? 8 : 0],
                     *(const v8bf*)&ldsB[buf][r][half ? 24 : 16]);
    }
#pragma unroll
    for (int mt = 0; mt < 2; ++mt)
#pragma unroll
      for (int nt = 0; nt < 2; ++nt)
        acc[mt][nt] = __builtin_amdgcn_wmma_f32_16x16x32_bf16(
            false, af[mt], false, bfv[nt], (short)0, acc[mt][nt], false, false);
    buf ^= 1;
  }

  // ---- epilogue ----
#pragma unroll
  for (int mt = 0; mt < 2; ++mt) {
    const int mTile = blockIdx.x * 4 + mW * 2 + mt;
    if (mTile >= N_TOK / 16) continue;          // guard (729 M tiles)
#pragma unroll
    for (int nt = 0; nt < 2; ++nt) {
      const int col = nBase + (nW * 2 + nt) * 16 + l16;
      const float bv_ = bias[col];
      if (mode == 0) {
        float* o = (float*)out;
#pragma unroll
        for (int i = 0; i < 8; ++i) {
          int m = mTile * 16 + i + half * 8;
          o[(size_t)m * DIM + col] = acc[mt][nt][i] + bv_;
        }
      } else {
        bf16* o = (bf16*)out;
        const int h = col / HDIM, d = col % HDIM;
#pragma unroll
        for (int i = 0; i < 8; ++i) {
          int m = mTile * 16 + i + half * 8;
          int b = m / SEQ, s = m % SEQ;
          int bh = b * HEADS + h;
          size_t idx = (mode == 1) ? ((size_t)bh * SP + s) * DP + d
                                   : ((size_t)bh * DP + d) * SP + s;
          o[idx] = (bf16)(acc[mt][nt][i] + bv_);
        }
      }
    }
  }
#undef STAGE
#undef ASYNC_WAIT
}

// ---------------------------------------------------------------------------
// Flash attention. One wave per (bh, qTile). Qp/Kp: [bh][SP][DP] bf16 (zero pad),
// Vt: [bh][DP][SP] bf16. Output: bf16 [token][DIM].
__global__ __launch_bounds__(256)
void attn_wmma_kernel(const bf16* __restrict__ Qp, const bf16* __restrict__ Kp,
                      const bf16* __restrict__ Vt, bf16* __restrict__ Oout) {
  __shared__ __align__(64) bf16 ldsP[8][16][32];    // per-wave P tile (16x32)

  const int lane = threadIdx.x & 31;
  const int wave = threadIdx.x >> 5;
  const int bh    = blockIdx.x;                     // 0..255
  const int qTile = blockIdx.y * 8 + wave;          // 0..47
  if (qTile >= QTILES) return;                      // wave-uniform, no barriers used
  const int half = lane >> 4;
  const int l16  = lane & 15;

  const bf16* Qb = Qp + (size_t)bh * SP * DP;
  const bf16* Kb = Kp + (size_t)bh * SP * DP;
  const bf16* Vb = Vt + (size_t)bh * DP * SP;

  // Q A-fragments for the 3 K-steps (Dh padded 72->96), kept resident
  v16bf aQ[3];
  {
    const bf16* qRow = Qb + (size_t)(qTile * 16 + l16) * DP;
#pragma unroll
    for (int tt = 0; tt < 3; ++tt) {
      v8bf lo = *(const v8bf*)(qRow + tt * 32 + (half ? 8  : 0));
      v8bf hi = *(const v8bf*)(qRow + tt * 32 + (half ? 24 : 16));
      aQ[tt] = cat8(lo, hi);
    }
  }

  // zero the K=16..31 half of the P tile once (A upper half stays zero)
#pragma unroll
  for (int i = 0; i < 8; ++i) ldsP[wave][i + half * 8][16 + l16] = (bf16)0.0f;

  float Mr[8], Lr[8];
  v8f Oacc[6];
#pragma unroll
  for (int i = 0; i < 8; ++i) { Mr[i] = -3.0e38f; Lr[i] = 0.0f; }
#pragma unroll
  for (int f = 0; f < 6; ++f) Oacc[f] = (v8f){};

  for (int j = 0; j < QTILES; ++j) {
    // ---- scores tile S = Q Kt : 3 WMMAs ----
    v8f c = {};
    const bf16* kRow = Kb + (size_t)(j * 16 + l16) * DP;
#pragma unroll
    for (int tt = 0; tt < 3; ++tt) {
      v16bf b = cat8(*(const v8bf*)(kRow + tt * 32 + (half ? 8 : 0)),
                     *(const v8bf*)(kRow + tt * 32 + (half ? 24 : 16)));
      c = __builtin_amdgcn_wmma_f32_16x16x32_bf16(false, aQ[tt], false, b, (short)0,
                                                  c, false, false);
    }
    const bool keyValid = (j * 16 + l16) < SEQ;     // per-lane key mask
#pragma unroll
    for (int i = 0; i < 8; ++i) c[i] = keyValid ? c[i] * ATTN_SCALE : -3.0e38f;

    // ---- online softmax per row (row group = 16 lanes of this half) ----
#pragma unroll
    for (int i = 0; i < 8; ++i) {
      float m = c[i];
      m = fmaxf(m, SWZ_XOR(m, 1)); m = fmaxf(m, SWZ_XOR(m, 2));
      m = fmaxf(m, SWZ_XOR(m, 4)); m = fmaxf(m, SWZ_XOR(m, 8));
      float newM = fmaxf(Mr[i], m);
      float corr = __expf(Mr[i] - newM);
      float pv   = __expf(c[i] - newM);
      float s = pv;
      s += SWZ_XOR(s, 1); s += SWZ_XOR(s, 2); s += SWZ_XOR(s, 4); s += SWZ_XOR(s, 8);
      Lr[i] = Lr[i] * corr + s;
      Mr[i] = newM;
#pragma unroll
      for (int f = 0; f < 6; ++f) Oacc[f][i] *= corr;
      ldsP[wave][i + half * 8][l16] = (bf16)pv;     // C/D layout -> LDS
    }
    __asm__ volatile("s_wait_dscnt 0x0" ::: "memory");

    // reload P in A-fragment layout (lane = row, K halves per lane group)
    v16bf pa = *(const v16bf*)&ldsP[wave][l16][half ? 16 : 0];

    // ---- O += P @ V : 6 WMMAs over feature tiles (Vt rows contiguous in key)
    // upper K half of P is zero, so clamp the lane>=16 key offset at the edge
    const int keyOff = j * 16 + ((half && j < QTILES - 1) ? 16 : 0);
#pragma unroll
    for (int f = 0; f < 6; ++f) {
      v16bf bv = *(const v16bf*)(Vb + (size_t)(f * 16 + l16) * SP + keyOff);
      Oacc[f] = __builtin_amdgcn_wmma_f32_16x16x32_bf16(false, pa, false, bv,
                                                        (short)0, Oacc[f],
                                                        false, false);
    }
  }

  // ---- epilogue: O / L, scatter to [token][DIM] (skip d>=72 pad, s>=729) ----
  const int b = bh >> 4, h = bh & 15;
#pragma unroll
  for (int f = 0; f < 5; ++f) {                     // f=5 is entirely pad (80..95)
    const int d = f * 16 + l16;
    if (d >= HDIM) continue;
#pragma unroll
    for (int i = 0; i < 8; ++i) {
      int s = qTile * 16 + i + half * 8;
      if (s < SEQ)
        Oout[((size_t)(b * SEQ + s)) * DIM + h * HDIM + d] = (bf16)(Oacc[f][i] / Lr[i]);
    }
  }
}

// ---------------------------------------------------------------------------
extern "C" void kernel_launch(void* const* d_in, const int* in_sizes, int n_in,
                              void* d_out, int out_size, void* d_ws, size_t ws_size,
                              hipStream_t stream) {
  const float* hs = (const float*)d_in[0];
  const float* wq = (const float*)d_in[1];
  const float* bq = (const float*)d_in[2];
  const float* wk = (const float*)d_in[3];
  const float* bk = (const float*)d_in[4];
  const float* wv = (const float*)d_in[5];
  const float* bv = (const float*)d_in[6];
  const float* wo = (const float*)d_in[7];
  const float* bo = (const float*)d_in[8];

  char* ws = (char*)d_ws;
  size_t off = 0;
  auto take = [&](size_t bytes) { void* p = ws + off; off += (bytes + 255) & ~(size_t)255; return p; };

  const size_t HS_B  = (size_t)N_TOK * DIM * sizeof(bf16);             // 26.9 MB
  const size_t W_B   = (size_t)DIM * DIM * sizeof(bf16);               //  2.65 MB
  const size_t QK_B  = (size_t)BATCH * HEADS * SP * DP * sizeof(bf16); // 36.2 MB

  bf16* hsb = (bf16*)take(HS_B);
  bf16* wqT = (bf16*)take(W_B);
  bf16* wkT = (bf16*)take(W_B);
  bf16* wvT = (bf16*)take(W_B);
  bf16* woT = (bf16*)take(W_B);
  bf16* Qp  = (bf16*)take(QK_B);
  bf16* Kp  = (bf16*)take(QK_B);
  bf16* Vt  = (bf16*)take(QK_B);
  bf16* AO  = (bf16*)take(HS_B);
  // total ~173 MB of workspace

  // zero the padded Q/K/V buffers (contiguous) so pad features multiply as 0
  (void)hipMemsetAsync(Qp, 0, 3 * QK_B, stream);

  // fp32 -> bf16 conversions
  cvt_bf16_kernel<<<(N_TOK * DIM + 255) / 256, 256, 0, stream>>>(hs, hsb, N_TOK * DIM);
  cvt_wT_kernel<<<(DIM * DIM) / 256, 256, 0, stream>>>(wq, wqT);
  cvt_wT_kernel<<<(DIM * DIM) / 256, 256, 0, stream>>>(wk, wkT);
  cvt_wT_kernel<<<(DIM * DIM) / 256, 256, 0, stream>>>(wv, wvT);
  cvt_wT_kernel<<<(DIM * DIM) / 256, 256, 0, stream>>>(wo, woT);

  // QKV projections (WMMA) with head-layout scatter epilogues
  dim3 gemmGrid((729 + 3) / 4, DIM / 128);   // 183 x 9, 8 waves/block
  gemm_wmma_kernel<<<gemmGrid, 256, 0, stream>>>(hsb, wqT, bq, Qp, 1);
  gemm_wmma_kernel<<<gemmGrid, 256, 0, stream>>>(hsb, wkT, bk, Kp, 1);
  gemm_wmma_kernel<<<gemmGrid, 256, 0, stream>>>(hsb, wvT, bv, Vt, 2);

  // flash attention (WMMA scores + WMMA P@V)
  attn_wmma_kernel<<<dim3(BATCH * HEADS, (QTILES + 7) / 8), 256, 0, stream>>>(Qp, Kp, Vt, AO);

  // output projection (WMMA), f32 + bias straight to d_out
  gemm_wmma_kernel<<<gemmGrid, 256, 0, stream>>>(AO, woT, bo, d_out, 0);
}
